// TransformerBlock_16071767622381
// MI455X (gfx1250) — compile-verified, matrix operand order checked by reading, not run on hardware
//
#include <hip/hip_runtime.h>

// ---------------------------------------------------------------------------
// Types (clang ext vectors only; trivially-copyable for unions)
// ---------------------------------------------------------------------------
typedef __attribute__((ext_vector_type(16))) _Float16 v16h;
typedef __attribute__((ext_vector_type(8)))  float    v8f;
typedef __attribute__((ext_vector_type(4)))  unsigned int u32x4;
typedef __attribute__((ext_vector_type(8)))  int      i32x8;
typedef __attribute__((ext_vector_type(4)))  int      i32x4;
typedef __attribute__((ext_vector_type(4)))  float    f32x4;

// Problem constants (fixed by the reference)
#define BB   2
#define TT   2048
#define DD   1024
#define HH   16
#define HD   64
#define MR   (BB * TT)      // 4096 rows

// Tensor Data Mover availability (ROCm 7.2: 5-arg builtin; amdgpu-toolchain
// therock headers present: 6-arg builtin). Fallback: VGPR staging.
#if defined(__has_builtin)
#if __has_builtin(__builtin_amdgcn_tensor_load_to_lds)
#define HAVE_TDM 1
#endif
#endif

// ---------------------------------------------------------------------------
// WMMA helpers
// ---------------------------------------------------------------------------
__device__ __forceinline__ v8f wmma32(v16h a, v16h b, v8f c) {
    // D = A(16x32 f16) * B(32x16 f16) + C(16x16 f32)
    return __builtin_amdgcn_wmma_f32_16x16x32_f16(
        false, a, false, b, (short)0, c, false, false);
}

// Load one 16-half fragment row: per the CDNA5 16-bit A/B layout, lanes 0-15
// hold K = {0..7, 16..23}, lanes 16-31 hold K = {8..15, 24..31}. With a
// contiguous 32-half row, that is uint4 chunks {0,2} (lo) / {1,3} (hi).
__device__ __forceinline__ v16h load_frag(const _Float16* p, int hi) {
    union { u32x4 q[2]; v16h h; } f;
    const u32x4* q = (const u32x4*)p;
    f.q[0] = q[hi ? 1 : 0];
    f.q[1] = q[hi ? 3 : 2];
    return f.h;
}

__device__ __forceinline__ float gelu_f(float x) {
    float c = 0.7978845608028654f * (1.0f + 0.044715f * x * x);
    float u = x * c;
    return 0.5f * x * (1.0f + __fdividef(u, 1.0f + fabsf(u)));
}

#ifdef HAVE_TDM
// ---------------------------------------------------------------------------
// TDM: DMA a 2D f16 tile [tileRows x tileK] from a row-major tensor with row
// length rowLen (elements) into contiguous LDS (tileRows rows of tileK halves).
// D# layout per CDNA5 ISA 8.3/8.4: group0 = {count, lds_addr, global_addr,
// type=2}; group1 = {data_size=2B, tensor dims, tile dims, dim0 stride}.
// ---------------------------------------------------------------------------
__device__ __forceinline__ void tdm_load_tile_f16(
    const _Float16* gptr, unsigned int lds_byte_addr,
    int rowLen, int numRows, int tileK, int tileRows)
{
    unsigned long long ga = (unsigned long long)(uintptr_t)gptr;
    u32x4 g0;
    g0.x = 1u;                                              // count=1 (valid)
    g0.y = lds_byte_addr;                                   // lds_addr
    g0.z = (unsigned int)(ga & 0xFFFFFFFFu);                // global_addr lo
    g0.w = (unsigned int)((ga >> 32) & 0x01FFFFFFu)         // global_addr hi
         | (2u << 30);                                      // type=2 (image)
    i32x8 g1;
    g1[0] = (1 << 16);                                      // data_size=1 (2B)
    g1[1] = (rowLen & 0xFFFF) << 16;                        // tensor_dim0 lo
    g1[2] = ((rowLen >> 16) & 0xFFFF)                       // tensor_dim0 hi
          | ((numRows & 0xFFFF) << 16);                     // tensor_dim1 lo
    g1[3] = ((numRows >> 16) & 0xFFFF)                      // tensor_dim1 hi
          | ((tileK & 0xFFFF) << 16);                       // tile_dim0
    g1[4] = (tileRows & 0xFFFF);                            // tile_dim1 (dim2=0)
    g1[5] = rowLen;                                         // dim0_stride lo
    g1[6] = 0;                                              // stride hi / dim1_stride
    g1[7] = 0;
    i32x4 z4 = {0, 0, 0, 0};
#if __has_include(<hip/amd_detail/amd_gfx1250_TDM.h>)
    i32x8 z8 = {0, 0, 0, 0, 0, 0, 0, 0};
    __builtin_amdgcn_tensor_load_to_lds(g0, g1, z4, z4, z8, 0);
#else
    __builtin_amdgcn_tensor_load_to_lds(g0, g1, z4, z4, 0);
#endif
}
#endif // HAVE_TDM

// ---------------------------------------------------------------------------
// Weight convert + transpose: in f32 [K,N] -> out f16 [N,K].
// 32x32 tiles, 256 threads (32 wide x 8 deep). K,N multiples of 32.
// ---------------------------------------------------------------------------
__global__ __launch_bounds__(256) void cvtT_kernel(const float* __restrict__ in,
                                                   _Float16* __restrict__ out,
                                                   int K, int N) {
    __shared__ _Float16 tile[32][33];
    int bn = blockIdx.x * 32;          // N offset
    int bk = blockIdx.y * 32;          // K offset
    int tx = threadIdx.x & 31, ty = threadIdx.x >> 5;   // ty: 0..7
#pragma unroll
    for (int i = 0; i < 32; i += 8)
        tile[ty + i][tx] = (_Float16)in[(size_t)(bk + ty + i) * N + bn + tx];
    __syncthreads();
    // out[n][k] = tile[k - bk][n - bn]; write coalesced along k
#pragma unroll
    for (int i = 0; i < 32; i += 8)
        out[(size_t)(bn + ty + i) * K + bk + tx] = tile[tx][ty + i];
}

// ---------------------------------------------------------------------------
// LayerNorm (one row per block, D=1024, 256 threads x 4 elems), f16 output
// ---------------------------------------------------------------------------
__global__ __launch_bounds__(256) void ln_f16_kernel(const float* __restrict__ x,
                                                     const float* __restrict__ w,
                                                     const float* __restrict__ b,
                                                     _Float16* __restrict__ out) {
    int row = blockIdx.x;
    int t = threadIdx.x;
    const f32x4* xr = (const f32x4*)(x + (size_t)row * DD);
    f32x4 v = xr[t];
    float s  = v.x + v.y + v.z + v.w;
    float s2 = v.x * v.x + v.y * v.y + v.z * v.z + v.w * v.w;

    __shared__ float rs[256];
    __shared__ float rq[256];
    rs[t] = s; rq[t] = s2;
    __syncthreads();
#pragma unroll
    for (int off = 128; off > 0; off >>= 1) {
        if (t < off) { rs[t] += rs[t + off]; rq[t] += rq[t + off]; }
        __syncthreads();
    }
    float mean = rs[0] * (1.0f / DD);
    float var  = rq[0] * (1.0f / DD) - mean * mean;
    float rstd = rsqrtf(var + 1e-5f);

    f32x4 wv = ((const f32x4*)w)[t];
    f32x4 bv = ((const f32x4*)b)[t];
    _Float16* orow = out + (size_t)row * DD;
    orow[4 * t + 0] = (_Float16)((v.x - mean) * rstd * wv.x + bv.x);
    orow[4 * t + 1] = (_Float16)((v.y - mean) * rstd * wv.y + bv.y);
    orow[4 * t + 2] = (_Float16)((v.z - mean) * rstd * wv.z + bv.z);
    orow[4 * t + 3] = (_Float16)((v.w - mean) * rstd * wv.w + bv.w);
}

// ---------------------------------------------------------------------------
// Tiled WMMA GEMM:  C[M,N] = A[M,K] @ WT[N,K]^T  (both f16, f32 accum)
// 256 threads = 8 waves; tile 128x128; each wave owns 32x64 (2x4 wmma tiles).
// Double-buffered LDS (32 KB). Tile staging uses the Tensor Data Mover when
// available (wave 0 issues 2 descriptors per k-step, waits TENSORcnt before
// the publishing barrier); otherwise coalesced b128 VGPR staging.
// Epilogue: optional residual add (f32), optional GELU, f32 or f16 store.
// ---------------------------------------------------------------------------
#define BM 128
#define BN 128
#define BK 32

__global__ __launch_bounds__(256) void gemm_wmma_kernel(
    const _Float16* __restrict__ A, const _Float16* __restrict__ WT,
    float* outF32, _Float16* outF16, const float* res,
    int M, int N, int K, int applyGelu) {

    __shared__ alignas(16) _Float16 As[2][BM][BK];   // 16 KB
    __shared__ alignas(16) _Float16 Bt[2][BN][BK];   // 16 KB ([n][k])

    int tid  = threadIdx.x;
    int lane = tid & 31, wave = tid >> 5;
    int hi   = lane >> 4, ln  = lane & 15;
    int wy   = wave >> 1, wx  = wave & 1;            // 4 waves in M, 2 in N
    int bm0  = blockIdx.y * BM, bn0 = blockIdx.x * BN;

    v8f acc[2][4];
    v8f zero = {0.f, 0.f, 0.f, 0.f, 0.f, 0.f, 0.f, 0.f};
#pragma unroll
    for (int mf = 0; mf < 2; ++mf)
#pragma unroll
        for (int nf = 0; nf < 4; ++nf) acc[mf][nf] = zero;

#ifdef HAVE_TDM
    bool w0 = (wave == 0);
    unsigned int ldsA0 = (unsigned int)(uintptr_t)&As[0][0][0];
    unsigned int ldsA1 = (unsigned int)(uintptr_t)&As[1][0][0];
    unsigned int ldsB0 = (unsigned int)(uintptr_t)&Bt[0][0][0];
    unsigned int ldsB1 = (unsigned int)(uintptr_t)&Bt[1][0][0];
    auto stage = [&](int k0, int buf) {
        if (w0) {
            tdm_load_tile_f16(A  + (size_t)bm0 * K + k0, buf ? ldsA1 : ldsA0,
                              K, M, BK, BM);
            tdm_load_tile_f16(WT + (size_t)bn0 * K + k0, buf ? ldsB1 : ldsB0,
                              K, N, BK, BN);
        }
    };
    auto stage_wait = [&]() {
        if (w0) __builtin_amdgcn_s_wait_tensorcnt((short)0);
    };
#else
    int crow = tid >> 1, ccol = (tid & 1) * 16;      // coop-copy mapping
    auto stage = [&](int k0, int buf) {
        const u32x4* ag = (const u32x4*)(A  + (size_t)(bm0 + crow) * K + k0 + ccol);
        const u32x4* bg = (const u32x4*)(WT + (size_t)(bn0 + crow) * K + k0 + ccol);
        *(u32x4*)&As[buf][crow][ccol]     = ag[0];
        *(u32x4*)&As[buf][crow][ccol + 8] = ag[1];
        *(u32x4*)&Bt[buf][crow][ccol]     = bg[0];
        *(u32x4*)&Bt[buf][crow][ccol + 8] = bg[1];
    };
    auto stage_wait = [&]() {};
#endif

    stage(0, 0);
    int nk = K / BK;
    int p = 0;
    for (int kb = 0; kb < nk; ++kb) {
        stage_wait();                 // issuing wave drains TENSORcnt
        __syncthreads();              // publish buffer p to all waves
        if (kb + 1 < nk) stage((kb + 1) * BK, p ^ 1);

        v16h af[2];
        af[0] = load_frag(&As[p][wy * 32 + ln][0], hi);
        af[1] = load_frag(&As[p][wy * 32 + 16 + ln][0], hi);
#pragma unroll
        for (int nf = 0; nf < 4; ++nf) {
            v16h bf = load_frag(&Bt[p][wx * 64 + nf * 16 + ln][0], hi);
            acc[0][nf] = wmma32(af[0], bf, acc[0][nf]);
            acc[1][nf] = wmma32(af[1], bf, acc[1][nf]);
        }
        p ^= 1;
    }

    bool hasRes = (res != nullptr);
#pragma unroll
    for (int mf = 0; mf < 2; ++mf)
#pragma unroll
        for (int nf = 0; nf < 4; ++nf) {
            int rowb = bm0 + wy * 32 + mf * 16 + hi * 8;
            int colg = bn0 + wx * 64 + nf * 16 + ln;
#pragma unroll
            for (int r = 0; r < 8; ++r) {
                size_t idx = (size_t)(rowb + r) * N + colg;
                float v = acc[mf][nf][r];
                if (hasRes) v += res[idx];
                if (applyGelu) v = gelu_f(v);
                if (outF32) outF32[idx] = v;
                else        outF16[idx] = (_Float16)v;
            }
        }
}

// ---------------------------------------------------------------------------
// Flash attention: grid (T/64, B*H), 128 threads = 4 waves.
// Each wave owns 16 q-rows; kv processed in 32-key blocks staged in LDS.
// K staged row-major (K-frag rows are contiguous); V staged transposed
// (hd-major) so the P@V B-fragment is also two contiguous ds_load_b128.
// qkv layout: row (b*T + t), cols [q: h*64+d | k: 1024+h*64+d | v: 2048+...]
// ---------------------------------------------------------------------------
__global__ __launch_bounds__(128) void attn_kernel(const _Float16* __restrict__ qkv,
                                                   _Float16* __restrict__ o) {
    const int NC = 3 * DD;
    int qt = blockIdx.x;                 // q tile (64 rows)
    int bh = blockIdx.y;
    int bb = bh >> 4, h = bh & 15;
    int tid = threadIdx.x;
    int lane = tid & 31, w = tid >> 5;
    int hi = lane >> 4, ln = lane & 15;

    __shared__ alignas(16) _Float16 Kt[32][64];       // 4 KB  [key][hd]
    __shared__ alignas(16) _Float16 Vtt[64][32];      // 4 KB  [hd][key]
    __shared__ float    Sb[4][16][32];                // 8 KB
    __shared__ alignas(16) _Float16 Pb[4][16][32];    // 4 KB
    __shared__ float Mr[4][16], Lr[4][16], Ar[4][16];

    size_t rowbase = (size_t)bb * TT;
    int q0 = qt * 64 + w * 16;

    // Q fragments (pre-scaled by 1/sqrt(hd) = 0.125, exact in f16)
    v16h qf[2];
    {
        const _Float16* qp = qkv + (rowbase + q0 + ln) * NC + h * HD;
#pragma unroll
        for (int kk = 0; kk < 2; ++kk) {
            union { u32x4 q[2]; v16h h; } f;
            const u32x4* p = (const u32x4*)(qp + kk * 32);
            f.q[0] = p[hi ? 1 : 0];
            f.q[1] = p[hi ? 3 : 2];
#pragma unroll
            for (int i = 0; i < 16; ++i) f.h[i] = f.h[i] * (_Float16)0.125f;
            qf[kk] = f.h;
        }
    }

    v8f of[4];
    v8f zero = {0.f, 0.f, 0.f, 0.f, 0.f, 0.f, 0.f, 0.f};
#pragma unroll
    for (int nf = 0; nf < 4; ++nf) of[nf] = zero;
    if (lane < 16) { Mr[w][lane] = -__builtin_inff(); Lr[w][lane] = 0.f; }

    int jend = qt * 64 + 64;             // uniform across the block (causal)
    for (int j0 = 0; j0 < jend; j0 += 32) {
        // cooperatively stage K (row-major) and V (transposed)
        {
            int kr = tid >> 2, c0 = (tid & 3) * 16;
            const u32x4* kg = (const u32x4*)(qkv + (rowbase + j0 + kr) * NC + DD + h * HD + c0);
            *(u32x4*)&Kt[kr][c0]     = kg[0];
            *(u32x4*)&Kt[kr][c0 + 8] = kg[1];
            const u32x4* vg = (const u32x4*)(qkv + (rowbase + j0 + kr) * NC + 2 * DD + h * HD + c0);
            union { u32x4 q; _Float16 hh[8]; } a, b2;
            a.q = vg[0]; b2.q = vg[1];
#pragma unroll
            for (int i = 0; i < 8; ++i) {
                Vtt[c0 + i][kr]     = a.hh[i];
                Vtt[c0 + 8 + i][kr] = b2.hh[i];
            }
        }
        __syncthreads();

        // S = Q @ K^T for 2 key tiles of 16; contraction over hd=64 (2 steps)
#pragma unroll
        for (int nt = 0; nt < 2; ++nt) {
            v8f s = zero;
            v16h kf0 = load_frag(&Kt[nt * 16 + ln][0],  hi);
            v16h kf1 = load_frag(&Kt[nt * 16 + ln][32], hi);
            s = wmma32(qf[0], kf0, s);
            s = wmma32(qf[1], kf1, s);
#pragma unroll
            for (int r = 0; r < 8; ++r) Sb[w][hi * 8 + r][nt * 16 + ln] = s[r];
        }

        // online softmax update (one lane per q-row; wave-private LDS)
        if (lane < 16) {
            int row = lane;
            int qrow = q0 + row;
            float mo = Mr[w][row], lo = Lr[w][row];
            float mx = mo;
            for (int c = 0; c < 32; ++c)
                if (j0 + c <= qrow) mx = fmaxf(mx, Sb[w][row][c]);
            float alpha = (mo > -__builtin_inff()) ? __expf(mo - mx) : 0.f;
            float ls = 0.f;
            for (int c = 0; c < 32; ++c) {
                float p = 0.f;
                if (j0 + c <= qrow && mx > -__builtin_inff()) {
                    p = __expf(Sb[w][row][c] - mx);
                    ls += p;
                }
                Pb[w][row][c] = (_Float16)p;
            }
            Mr[w][row] = mx;
            Lr[w][row] = lo * alpha + ls;
            Ar[w][row] = alpha;
        }

        // rescale O accumulators, then O += P @ V   (EXEC fully converged here)
        float av[8];
#pragma unroll
        for (int r = 0; r < 8; ++r) av[r] = Ar[w][hi * 8 + r];
#pragma unroll
        for (int nf = 0; nf < 4; ++nf)
#pragma unroll
            for (int r = 0; r < 8; ++r) of[nf][r] *= av[r];

        v16h pf = load_frag(&Pb[w][ln][0], hi);
#pragma unroll
        for (int nf = 0; nf < 4; ++nf) {
            v16h vf = load_frag(&Vtt[nf * 16 + ln][0], hi);
            of[nf] = wmma32(pf, vf, of[nf]);
        }
        __syncthreads();   // protect Kt/Vtt before next block overwrites them
    }

    // finalize: divide by running denom (ref adds 1e-6) and store f16 output
    if (lane < 16) Ar[w][lane] = 1.0f / (Lr[w][lane] + 1e-6f);
    float iv[8];
#pragma unroll
    for (int r = 0; r < 8; ++r) iv[r] = Ar[w][hi * 8 + r];
#pragma unroll
    for (int nf = 0; nf < 4; ++nf)
#pragma unroll
        for (int r = 0; r < 8; ++r) {
            int row = q0 + hi * 8 + r;
            int col = h * HD + nf * 16 + ln;
            o[(rowbase + row) * DD + col] = (_Float16)(of[nf][r] * iv[r]);
        }
}

// ---------------------------------------------------------------------------
// Host launcher.  Workspace layout (64 MiB total):
//   [0, 8M)    h16   (LN output, reused for LN2)
//   [8M, 32M)  qkv16 (24 MiB)
//   [32M, 40M) o16   (8 MiB)
//   g16 aliases [8M, 40M) (32 MiB; qkv16/o16 are dead when FF1 writes it)
//   [40M, 64M) transposed f16 weights
// ---------------------------------------------------------------------------
extern "C" void kernel_launch(void* const* d_in, const int* in_sizes, int n_in,
                              void* d_out, int out_size, void* d_ws, size_t ws_size,
                              hipStream_t stream) {
    const float* x     = (const float*)d_in[0];
    const float* ln1_w = (const float*)d_in[1];
    const float* ln1_b = (const float*)d_in[2];
    const float* ln2_w = (const float*)d_in[3];
    const float* ln2_b = (const float*)d_in[4];
    const float* w_qkv = (const float*)d_in[5];
    const float* w_prj = (const float*)d_in[6];
    const float* w_ff1 = (const float*)d_in[7];
    const float* w_ff2 = (const float*)d_in[8];
    float* out = (float*)d_out;

    unsigned char* ws = (unsigned char*)d_ws;
    _Float16* h16    = (_Float16*)(ws);                                   // 8 MiB
    _Float16* qkv16  = (_Float16*)(ws + (size_t)8u * 1024 * 1024);        // 24 MiB
    _Float16* o16    = (_Float16*)(ws + (size_t)32u * 1024 * 1024);       // 8 MiB
    _Float16* g16    = (_Float16*)(ws + (size_t)8u * 1024 * 1024);        // 32 MiB (alias)
    _Float16* wqkvT  = (_Float16*)(ws + (size_t)40u * 1024 * 1024);       // 6 MiB
    _Float16* wprjT  = (_Float16*)(ws + (size_t)46u * 1024 * 1024);       // 2 MiB
    _Float16* wff1T  = (_Float16*)(ws + (size_t)48u * 1024 * 1024);       // 8 MiB
    _Float16* wff2T  = (_Float16*)(ws + (size_t)56u * 1024 * 1024);       // 8 MiB

    // weight convert + transpose: [K,N] f32 -> [N,K] f16
    cvtT_kernel<<<dim3((3 * DD) / 32, DD / 32), 256, 0, stream>>>(w_qkv, wqkvT, DD, 3 * DD);
    cvtT_kernel<<<dim3(DD / 32, DD / 32),       256, 0, stream>>>(w_prj, wprjT, DD, DD);
    cvtT_kernel<<<dim3((4 * DD) / 32, DD / 32), 256, 0, stream>>>(w_ff1, wff1T, DD, 4 * DD);
    cvtT_kernel<<<dim3(DD / 32, (4 * DD) / 32), 256, 0, stream>>>(w_ff2, wff2T, 4 * DD, DD);

    // ---- attention sublayer ----
    ln_f16_kernel<<<dim3(MR), 256, 0, stream>>>(x, ln1_w, ln1_b, h16);

    gemm_wmma_kernel<<<dim3((3 * DD) / BN, MR / BM), 256, 0, stream>>>(
        h16, wqkvT, nullptr, qkv16, nullptr, MR, 3 * DD, DD, 0);

    attn_kernel<<<dim3(TT / 64, BB * HH), 128, 0, stream>>>(qkv16, o16);

    // x1 = x + o @ w_proj   (written to d_out)
    gemm_wmma_kernel<<<dim3(DD / BN, MR / BM), 256, 0, stream>>>(
        o16, wprjT, out, nullptr, x, MR, DD, DD, 0);

    // ---- MLP sublayer ----
    ln_f16_kernel<<<dim3(MR), 256, 0, stream>>>(out, ln2_w, ln2_b, h16);

    gemm_wmma_kernel<<<dim3((4 * DD) / BN, MR / BM), 256, 0, stream>>>(
        h16, wff1T, nullptr, g16, nullptr, MR, 4 * DD, DD, 1 /*gelu*/);

    // out = x1 + gelu(h @ ff1) @ ff2   (residual read+write of d_out is safe:
    // each element is read then written by exactly one thread)
    gemm_wmma_kernel<<<dim3(DD / BN, MR / BM), 256, 0, stream>>>(
        g16, wff2T, out, nullptr, out, MR, DD, 4 * DD, 0);
}
